// MessagePassing_14061722927244
// MI455X (gfx1250) — matrix operand order repaired and checked by reading, not compiled, not run on hardware
//
#include <hip/hip_runtime.h>
#include <hip/hip_bf16.h>
#include <string.h>

// ---------------------------------------------------------------------------
// MessagePassing GNN layer for MI455X (gfx1250, wave32, WMMA).
//   out = x @ Wu^T + (adj @ x) @ Wa^T + (bu + ba)
// adj is a 0/1 SET-semantics symmetric adjacency built from edge_index.
//
// Pipeline (all on `stream`, graph-capture safe):
//   k0: zero bitmap+deg
//   k1: swizzle Wa/Wu into bf16 hi/lo B-fragment layout; bsum = bu+ba
//   k2: build bitmap via atomicOr (exact dedup) + per-row neighbor lists
//   k3: fused per-16-row-tile kernel: sparse aggregate -> LDS-staged
//       A-fragments -> bf16x3 split GEMM via v_wmma_f32_16x16x32_bf16
// ---------------------------------------------------------------------------

typedef unsigned int u32;
typedef __attribute__((ext_vector_type(16))) __bf16 v16bf;
typedef __attribute__((ext_vector_type(8)))  float  v8f;
typedef __attribute__((ext_vector_type(8)))  u32    v8u;

#define NNODES  16384
#define DK      256      // D_IN == D_OUT
#define NWORDS  512      // bitmap u32 words per row (16384/32)
#define MAXD    256      // neighbor-list capacity per row (mean deg ~64)

// ---- bf16 split helpers (manual RNE, no reliance on __bf16 scalar ops) ----
__device__ __forceinline__ u32 f32_to_bf16_rne(float f) {
  u32 u = __builtin_bit_cast(u32, f);
  u32 r = u + 0x7fffu + ((u >> 16) & 1u);
  return r >> 16;
}
__device__ __forceinline__ float bf16_to_f32(u32 h) {
  return __builtin_bit_cast(float, h << 16);
}
// Pack (a,b) into hi-dword (bf16 of a|b) and lo-dword (bf16 residuals).
__device__ __forceinline__ void split2(float a, float b, u32& hi, u32& lo) {
  u32 ah = f32_to_bf16_rne(a);
  u32 bh = f32_to_bf16_rne(b);
  u32 al = f32_to_bf16_rne(a - bf16_to_f32(ah));
  u32 bl = f32_to_bf16_rne(b - bf16_to_f32(bh));
  hi = ah | (bh << 16);
  lo = al | (bl << 16);
}

__device__ __forceinline__ v8f wmma_bf16(v16bf a, v16bf b, v8f c) {
  // 8 args: (neg_a, A, neg_b, B, c_mod, C, reuse_a, reuse_b)
  return __builtin_amdgcn_wmma_f32_16x16x32_bf16(
      false, a, false, b, (short)0, c, false, false);
}

__device__ __forceinline__ v16bf load_frag_lds(const u32* p) {
  v8u u = *(const v8u*)p;                 // 2x ds_load_b128
  return __builtin_bit_cast(v16bf, u);
}
__device__ __forceinline__ v16bf load_frag_glb(const u32* p) {
  v8u u = *(const v8u*)p;                 // 2x global_load_b128 (coalesced)
  return __builtin_bit_cast(v16bf, u);
}

// A-fragment staging index: element (m, c) of a 16x256 tile -> swizzled
// dword slot in [kstep][lane][vgpr] order. (c is even; (c,c+1) share a dword.)
// 16-bit A layout (ISA 7.12.2): lanes0-15 hold K 0..7 (v0-3) & 16..23 (v4-7),
// lanes16-31 hold K 8..15 & 24..31, M = lane%16, even K in [15:0].
__device__ __forceinline__ int a_stage_idx(int m, int c) {
  int ks = c >> 5;
  int kk = c & 31;
  int Ld, v;
  if (kk < 8)       { Ld = m;      v = kk >> 1; }
  else if (kk < 16) { Ld = m + 16; v = (kk - 8) >> 1; }
  else if (kk < 24) { Ld = m;      v = 4 + ((kk - 16) >> 1); }
  else              { Ld = m + 16; v = 4 + ((kk - 24) >> 1); }
  return (ks * 32 + Ld) * 8 + v;
}

// ---------------------------------------------------------------------------
// k0: zero bitmap + degree counters (must run every launch: ws is not
// re-initialized between graph replays).
__global__ void zero_kernel(u32* __restrict__ p, unsigned long long n) {
  unsigned long long i = (unsigned long long)blockIdx.x * blockDim.x + threadIdx.x;
  unsigned long long stride = (unsigned long long)gridDim.x * blockDim.x;
  for (; i < n; i += stride) p[i] = 0u;
}

// ---------------------------------------------------------------------------
// k1: pre-swizzle W matrices into bf16 hi/lo B-fragments.
// B layout (16-bit, 32x16, 8 VGPRs): lane n=L&15, K-half = L>>4;
// VGPR v holds K = (v>>2)*16 + khalf*8 + 2(v&3) (lo half) and +1 (hi half), value W[n][k].
// Stored as dwords at ((nt*8+ks)*32+L)*8+v == tid, so GEMM loads are b128.
__global__ void prep_w_kernel(const float* __restrict__ Wa,
                              const float* __restrict__ Wu,
                              const float* __restrict__ bu,
                              const float* __restrict__ ba,
                              u32* __restrict__ waHi, u32* __restrict__ waLo,
                              u32* __restrict__ wuHi, u32* __restrict__ wuLo,
                              float* __restrict__ bsum) {
  int tid = blockIdx.x * blockDim.x + threadIdx.x;   // 0 .. 32767
  int v  = tid & 7;
  int L  = (tid >> 3) & 31;
  int ks = (tid >> 8) & 7;
  int nt = tid >> 11;
  int n = nt * 16 + (L & 15);
  int k = ks * 32 + (v >> 2) * 16 + (L >> 4) * 8 + 2 * (v & 3);
  u32 h, l;
  split2(Wa[n * DK + k], Wa[n * DK + k + 1], h, l);
  waHi[tid] = h; waLo[tid] = l;
  split2(Wu[n * DK + k], Wu[n * DK + k + 1], h, l);
  wuHi[tid] = h; wuLo[tid] = l;
  if (tid < DK) bsum[tid] = bu[tid] + ba[tid];
}

// ---------------------------------------------------------------------------
// k2: exact SET-semantics adjacency build. atomicOr's pre-op value arbitrates
// duplicates: exactly one inserter per (a,b) appends to the neighbor list.
__device__ __forceinline__ void adj_insert(int a, int b,
                                           u32* __restrict__ bitmap,
                                           u32* __restrict__ deg,
                                           int* __restrict__ nbr) {
  u32 wi = (u32)a * NWORDS + ((u32)b >> 5);
  u32 mask = 1u << (b & 31);
  u32 old = atomicOr(&bitmap[wi], mask);
  if (!(old & mask)) {
    u32 p = atomicAdd(&deg[a], 1u);
    if (p < MAXD) nbr[(unsigned long long)a * MAXD + p] = b;
  }
}

__global__ void build_edges_kernel(const int* __restrict__ ei, int E,
                                   u32* __restrict__ bitmap,
                                   u32* __restrict__ deg,
                                   int* __restrict__ nbr) {
  int e = blockIdx.x * blockDim.x + threadIdx.x;
  if (e >= E) return;
  int s = ei[e];
  int d = ei[E + e];
  if ((unsigned)s >= (unsigned)NNODES || (unsigned)d >= (unsigned)NNODES) return;
  adj_insert(s, d, bitmap, deg, nbr);
  adj_insert(d, s, bitmap, deg, nbr);
}

// ---------------------------------------------------------------------------
// k3: fused aggregate + bf16x3 split GEMM. One wave per block, one 16x256
// output tile per wave. 32KB static LDS. EXEC stays all-ones (uniform flow).
__global__ __launch_bounds__(32) void fused_tile_kernel(
    const float* __restrict__ x,
    const u32*   __restrict__ deg,
    const int*   __restrict__ nbr,
    const u32*   __restrict__ waHi, const u32* __restrict__ waLo,
    const u32*   __restrict__ wuHi, const u32* __restrict__ wuLo,
    const float* __restrict__ bsum,
    float*       __restrict__ out) {
  __shared__ u32 sMsgHi[2048];
  __shared__ u32 sMsgLo[2048];
  __shared__ u32 sXHi[2048];
  __shared__ u32 sXLo[2048];

  const int L  = threadIdx.x;          // lane (wave32)
  const int r0 = blockIdx.x * 16;      // first node row of this tile
  const int c0 = L * 8;                // 8 feature columns owned by this lane

  // ---- Phase A: sparse aggregation msg[m][c] = sum_{j in N(r0+m)} x[j][c],
  //      plus staging of both A-operand tiles (msg, x) as split-bf16 fragments.
  for (int m = 0; m < 16; ++m) {
    const int row = r0 + m;
    float acc[8];
#pragma unroll
    for (int q = 0; q < 8; ++q) acc[q] = 0.0f;

    u32 cnt = deg[row];                // wave-uniform scalar load
    if (cnt > MAXD) cnt = MAXD;
    for (u32 t = 0; t < cnt; ++t) {
      int j = nbr[(unsigned long long)row * MAXD + t];   // uniform
      const float4* xr = (const float4*)(x + (unsigned long long)j * DK + c0);
      float4 a = xr[0];
      float4 b = xr[1];
      acc[0] += a.x; acc[1] += a.y; acc[2] += a.z; acc[3] += a.w;
      acc[4] += b.x; acc[5] += b.y; acc[6] += b.z; acc[7] += b.w;
    }

    const float4* xs = (const float4*)(x + (unsigned long long)row * DK + c0);
    float4 xa = xs[0];
    float4 xb = xs[1];
    float xv[8] = {xa.x, xa.y, xa.z, xa.w, xb.x, xb.y, xb.z, xb.w};

#pragma unroll
    for (int t2 = 0; t2 < 4; ++t2) {
      const int c = c0 + 2 * t2;
      const int idx = a_stage_idx(m, c);
      u32 h, l;
      split2(acc[2 * t2], acc[2 * t2 + 1], h, l);
      sMsgHi[idx] = h; sMsgLo[idx] = l;
      split2(xv[2 * t2], xv[2 * t2 + 1], h, l);
      sXHi[idx] = h; sXLo[idx] = l;
    }
  }
  __syncthreads();   // single-wave WG: lowers to waits only

  // ---- Phase B: out_tile = msg @ Wa^T + x @ Wu^T + bsum, bf16x3 split.
  for (int nt = 0; nt < 16; ++nt) {
    v8f c = {};
    for (int ks = 0; ks < 8; ++ks) {
      const int aoff = (ks * 32 + L) * 8;
      v16bf amh = load_frag_lds(&sMsgHi[aoff]);
      v16bf aml = load_frag_lds(&sMsgLo[aoff]);
      v16bf axh = load_frag_lds(&sXHi[aoff]);
      v16bf axl = load_frag_lds(&sXLo[aoff]);
      const int boff = ((nt * 8 + ks) * 32 + L) * 8;
      v16bf bah = load_frag_glb(&waHi[boff]);
      v16bf bal = load_frag_glb(&waLo[boff]);
      v16bf buh = load_frag_glb(&wuHi[boff]);
      v16bf bul = load_frag_glb(&wuLo[boff]);
      // hi*hi + hi*lo + lo*hi  (drop lo*lo, ~2^-18 rel)
      c = wmma_bf16(amh, bah, c);
      c = wmma_bf16(amh, bal, c);
      c = wmma_bf16(aml, bah, c);
      c = wmma_bf16(axh, buh, c);
      c = wmma_bf16(axh, bul, c);
      c = wmma_bf16(axl, buh, c);
    }
    // C/D layout: VGPR r -> (M = r + 8*(L>=16), N = nt*16 + (L&15))
    const int nc = nt * 16 + (L & 15);
    const float bs = bsum[nc];
    const int mbase = r0 + ((L >> 4) << 3);
#pragma unroll
    for (int r = 0; r < 8; ++r) {
      out[(unsigned long long)(mbase + r) * DK + nc] = c[r] + bs;
    }
  }
}

// ---------------------------------------------------------------------------
extern "C" void kernel_launch(void* const* d_in, const int* in_sizes, int n_in,
                              void* d_out, int out_size, void* d_ws, size_t ws_size,
                              hipStream_t stream) {
  (void)n_in; (void)out_size; (void)ws_size;
  const float* x  = (const float*)d_in[0];
  const int*   ei = (const int*)d_in[1];
  const float* Wu = (const float*)d_in[2];
  const float* bu = (const float*)d_in[3];
  const float* Wa = (const float*)d_in[4];
  const float* ba = (const float*)d_in[5];
  float* out = (float*)d_out;
  const int E = in_sizes[1] / 2;

  // Workspace carve-up (~51 MB total).
  char* ws = (char*)d_ws;
  const unsigned long long BITMAP_B = (unsigned long long)NNODES * NWORDS * 4ull; // 32 MB
  const unsigned long long DEG_B    = (unsigned long long)NNODES * 4ull;          // 64 KB
  const unsigned long long NBR_B    = (unsigned long long)NNODES * MAXD * 4ull;   // 16 MB
  const unsigned long long WSWZ_B   = 32768ull * 4ull;                            // 128 KB each
  u32* bitmap = (u32*)ws;
  u32* deg    = (u32*)(ws + BITMAP_B);
  int* nbr    = (int*)(ws + BITMAP_B + DEG_B);
  unsigned long long o = BITMAP_B + DEG_B + NBR_B;
  u32* waHi = (u32*)(ws + o); o += WSWZ_B;
  u32* waLo = (u32*)(ws + o); o += WSWZ_B;
  u32* wuHi = (u32*)(ws + o); o += WSWZ_B;
  u32* wuLo = (u32*)(ws + o); o += WSWZ_B;
  float* bsum = (float*)(ws + o);

  const unsigned long long zero_words = (BITMAP_B + DEG_B) / 4ull;
  zero_kernel<<<2048, 256, 0, stream>>>(bitmap, zero_words);
  prep_w_kernel<<<128, 256, 0, stream>>>(Wa, Wu, bu, ba,
                                         waHi, waLo, wuHi, wuLo, bsum);
  build_edges_kernel<<<(E + 255) / 256, 256, 0, stream>>>(ei, E, bitmap, deg, nbr);
  fused_tile_kernel<<<NNODES / 16, 32, 0, stream>>>(x, deg, nbr,
                                                    waHi, waLo, wuHi, wuLo,
                                                    bsum, out);
}
